// MemoryAttention_23003844837491
// MI455X (gfx1250) — matrix operand order repaired and checked by reading, not compiled, not run on hardware
//
#include <hip/hip_runtime.h>
#include <hip/hip_fp16.h>

typedef __attribute__((ext_vector_type(16))) _Float16 v16h;
typedef __attribute__((ext_vector_type(8)))  float    v8f;

#define QS   512
#define BB   8
#define CC   1024
#define HH   16
#define HC   64
#define MM   1536
#define KSZ  512
#define TT   2048          // MM + KSZ
#define NROWS 4096         // QS * BB

// ---------------------------------------------------------------- helpers
__device__ inline v8f wmma_f16(v16h a, v16h b, v8f c) {
  return __builtin_amdgcn_wmma_f32_16x16x32_f16(
      false, a, false, b, (short)0, c, false, false);
}

// A-tile (16 x 32): lane = row (l&15); K chunks of 8 halves at k0+half*8 and k0+16+half*8
__device__ inline v16h load_a_tile(const _Float16* base, int ldk, int lane, int k0) {
  int row  = lane & 15;
  int half = lane >> 4;
  const _Float16* p = base + (size_t)row * ldk + k0 + half * 8;
  v16h a;
  *(float4*)&a       = *(const float4*)(p);
  *((float4*)&a + 1) = *(const float4*)(p + 16);
  return a;
}

// B-tile (32 x 16), NT layout in global memory: column n stored as row n of base
__device__ inline v16h load_b_tile(const _Float16* base, int ldk, int lane, int k0) {
  int col  = lane & 15;
  int half = lane >> 4;
  const _Float16* p = base + (size_t)col * ldk + k0 + half * 8;
  v16h b;
  *(float4*)&b       = *(const float4*)(p);
  *((float4*)&b + 1) = *(const float4*)(p + 16);
  return b;
}

// B-tile from an LDS-staged 64x32 slab (row = column of W, 32 halves contiguous)
__device__ inline v16h load_b_lds(const _Float16* base, int lane, int nt) {
  int col  = lane & 15;
  int half = lane >> 4;
  const _Float16* p = base + ((nt * 16 + col) * 32) + half * 8;
  v16h b;
  *(float4*)&b       = *(const float4*)(p);
  *((float4*)&b + 1) = *(const float4*)(p + 16);
  return b;
}

// gfx1250 async copy: 16 bytes global -> LDS, tracked by ASYNCcnt
__device__ inline void async_copy_b128(uint32_t lds_addr, const void* gptr) {
  uint64_t ga = (uint64_t)(uintptr_t)gptr;
  asm volatile("global_load_async_to_lds_b128 %0, %1, off"
               :: "v"(lds_addr), "v"(ga) : "memory");
}

// ---------------------------------------------------------------- kernels
__global__ __launch_bounds__(256)
void cvt_f32_f16(const float* __restrict__ src, _Float16* __restrict__ dst, int n) {
  int i = blockIdx.x * 256 + threadIdx.x;
  if (i < n) dst[i] = (_Float16)src[i];
}

// Y = A(N x 1024) * W(O x 1024)^T + bias.
// Block = 8 waves, covers 256 rows x 64 cols; W tile staged in LDS via async
// loads with double buffering so every wave shares one copy of the weights.
// mode 0: write f16 scattered into [b][h][s][d]   (Q/K/V projections)
// mode 1: write f32 row-major N x O               (output projection)
__global__ __launch_bounds__(256)
void gemm_nt(const _Float16* __restrict__ A,
             const _Float16* __restrict__ W,
             const float*    __restrict__ bias,
             _Float16* __restrict__ outQKV,
             float*    __restrict__ outF32,
             int mode)
{
  __shared__ _Float16 Bs[2][64 * 32];   // two 4KB W tiles (64 cols x 32 K)

  int tid  = threadIdx.x;
  int lane = tid & 31;
  int wv   = tid >> 5;
  int blockRow = blockIdx.x >> 4;       // 16 row-blocks of 256 rows
  int blockCol = blockIdx.x & 15;       // 16 col-blocks of 64 cols
  int row0 = blockRow * 256 + wv * 32;
  int col0 = blockCol * 64;

  // Cooperative copy mapping: 256 lanes x 16B = 4KB tile.
  int ccol = tid >> 2;                  // 0..63 : W row (output column)
  int ck   = (tid & 3) * 8;             // 0,8,16,24 : K chunk of 8 halves
  const _Float16* wlane = W + (size_t)(col0 + ccol) * CC + ck;
  uint32_t lslot = (uint32_t)(ccol * 32 + ck) * 2;
  uint32_t laddr0 = (uint32_t)(uintptr_t)(&Bs[0][0]) + lslot;
  uint32_t laddr1 = (uint32_t)(uintptr_t)(&Bs[1][0]) + lslot;

  v8f acc[2][4];
  for (int mt = 0; mt < 2; ++mt)
    for (int nt = 0; nt < 4; ++nt)
      for (int j = 0; j < 8; ++j) acc[mt][nt][j] = 0.f;

  // Prefetch first W tile into buffer 0
  async_copy_b128(laddr0, wlane);

  for (int it = 0; it < 32; ++it) {
    int k0  = it * 32;
    int cur = it & 1;

    if (it + 1 < 32) {
      // prefetch next tile into the other buffer, keep it outstanding
      async_copy_b128(cur ? laddr0 : laddr1, wlane + (k0 + 32));
      asm volatile("s_wait_asynccnt 1" ::: "memory");
    } else {
      asm volatile("s_wait_asynccnt 0" ::: "memory");
    }
    __syncthreads();                    // current buffer visible to all waves

    v16h a0 = load_a_tile(A + (size_t)row0 * CC,        CC, lane, k0);
    v16h a1 = load_a_tile(A + (size_t)(row0 + 16) * CC, CC, lane, k0);
    const _Float16* bbase = &Bs[cur][0];
    for (int nt = 0; nt < 4; ++nt) {
      v16h bt = load_b_lds(bbase, lane, nt);
      acc[0][nt] = wmma_f16(a0, bt, acc[0][nt]);
      acc[1][nt] = wmma_f16(a1, bt, acc[1][nt]);
    }
    __syncthreads();                    // done reading before it gets refilled
  }

  int half = lane >> 4, n = lane & 15;
  for (int mt = 0; mt < 2; ++mt)
    for (int nt = 0; nt < 4; ++nt)
      for (int j = 0; j < 8; ++j) {
        int r = row0 + mt * 16 + j + 8 * half;   // row in [0,4096)
        int o = col0 + nt * 16 + n;              // col in [0,1024)
        float v = acc[mt][nt][j] + bias[o];
        if (mode == 0) {
          int s = r >> 3, b = r & 7;             // r = s*B + b
          int h = o >> 6, d = o & 63;            // o = h*HC + d
          outQKV[(((size_t)(b * HH + h) * QS + s) * HC) + d] = (_Float16)v;
        } else {
          outF32[(size_t)r * CC + o] = v;
        }
      }
}

// Build combined K [b][h][t][d] and transposed V [b][h][d][t] (f16)
__global__ __launch_bounds__(256)
void build_kv(const float* __restrict__ mkv,       // (M,B,H,2*HC) f32
              const _Float16* __restrict__ K16,    // [b][h][s][d]
              const _Float16* __restrict__ V16,
              const int* __restrict__ mem_len,
              _Float16* __restrict__ Kc,
              _Float16* __restrict__ Vt)
{
  size_t i = (size_t)blockIdx.x * 256 + threadIdx.x;  // B*H*T*HC = 2^24
  int d = (int)(i & 63);
  int t = (int)((i >> 6) & (TT - 1));
  int h = (int)((i >> 17) & 15);
  int b = (int)(i >> 21);

  int L = mem_len[b];
  _Float16 kv, vv;
  if (t < L) {
    int tm = t < (MM - 1) ? t : (MM - 1);
    size_t base = (((size_t)tm * BB + b) * HH + h) * (2 * HC);
    kv = (_Float16)mkv[base + d];
    vv = (_Float16)mkv[base + HC + d];
  } else {
    int off = t - L;
    int idx = off < (KSZ - 1) ? off : (KSZ - 1);
    size_t base = ((size_t)(b * HH + h) * QS + idx) * HC + d;
    kv = K16[base];
    vv = V16[base];
  }
  Kc[((size_t)(b * HH + h) * TT + t) * HC + d] = kv;
  Vt[((size_t)(b * HH + h) * HC + d) * TT + t] = vv;
}

// Combined mask bytes [b][t][s]: 1 => score = -inf
__global__ __launch_bounds__(256)
void build_mask(const unsigned char* __restrict__ mask,   // (QS,KS,B) bool
                const int* __restrict__ mem_len,
                const int* __restrict__ pad,
                unsigned char* __restrict__ cmask)
{
  size_t i = (size_t)blockIdx.x * 256 + threadIdx.x;   // B*T*QS = 2^23
  int s = (int)(i & (QS - 1));
  int t = (int)((i >> 9) & (TT - 1));
  int b = (int)(i >> 20);

  int L = mem_len[b];
  unsigned char m;
  if (t < L) {
    m = 0;
  } else {
    int off = t - L;
    if (off >= pad[b]) m = 1;
    else {
      int idx = off < (KSZ - 1) ? off : (KSZ - 1);
      m = mask[((size_t)s * KSZ + idx) * BB + b];
    }
  }
  cmask[((size_t)b * TT + t) * QS + s] = m;
}

// Flash attention: one wave per (b, h, 16-row s-tile)
__global__ __launch_bounds__(32)
void attention(const _Float16* __restrict__ Q,     // [b][h][s][d]
               const _Float16* __restrict__ Kc,    // [b][h][t][d]
               const _Float16* __restrict__ Vt,    // [b][h][d][t]
               const unsigned char* __restrict__ cmask, // [b][t][s]
               _Float16* __restrict__ X)           // [s*B+b][h*64+d] f16
{
  __shared__ _Float16 ldsP[16 * 32];

  int bid = blockIdx.x;
  int st = bid & 31;
  int h  = (bid >> 5) & 15;
  int b  = bid >> 9;
  int s0 = st * 16;
  int lane = threadIdx.x;
  int half = lane >> 4, nidx = lane & 15;

  const _Float16* qbase = Q  + ((size_t)(b * HH + h) * QS + s0) * HC;
  const _Float16* kbase = Kc + ((size_t)(b * HH + h) * TT) * HC;
  const _Float16* vbase = Vt + ((size_t)(b * HH + h) * HC) * TT;
  const unsigned char* mbase = cmask + (size_t)b * TT * QS;

  v16h qa0 = load_a_tile(qbase, HC, lane, 0);
  v16h qa1 = load_a_tile(qbase, HC, lane, 32);

  v8f o_acc[4];
  float mrow[8], lrow[8];
  for (int dt = 0; dt < 4; ++dt)
    for (int j = 0; j < 8; ++j) o_acc[dt][j] = 0.f;
  for (int j = 0; j < 8; ++j) { mrow[j] = -INFINITY; lrow[j] = 0.f; }

  const float temp = 0.125f;   // 1/sqrt(HC)

  for (int t0 = 0; t0 < TT; t0 += 32) {
    // S = Q * Kc^T for two 16-wide t tiles
    v8f sc[2];
    for (int tt = 0; tt < 2; ++tt) {
      for (int j = 0; j < 8; ++j) sc[tt][j] = 0.f;
      const _Float16* kb = kbase + (size_t)(t0 + tt * 16) * HC;
      v16h b0 = load_b_tile(kb, HC, lane, 0);
      v16h b1 = load_b_tile(kb, HC, lane, 32);
      sc[tt] = wmma_f16(qa0, b0, sc[tt]);
      sc[tt] = wmma_f16(qa1, b1, sc[tt]);
    }

    for (int j = 0; j < 8; ++j) {
      int s = s0 + j + 8 * half;
      float v0, v1;
      {
        int t = t0 + nidx;
        float v = sc[0][j] * temp;
        if (mbase[(size_t)t * QS + s]) v = -INFINITY;
        v0 = v;
        t = t0 + 16 + nidx;
        v = sc[1][j] * temp;
        if (mbase[(size_t)t * QS + s]) v = -INFINITY;
        v1 = v;
      }
      float rm = fmaxf(v0, v1);
      for (int off = 1; off < 16; off <<= 1)
        rm = fmaxf(rm, __shfl_xor(rm, off, 32));
      float nm = fmaxf(mrow[j], rm);

      float p0 = (v0 == -INFINITY) ? 0.f : __expf(v0 - nm);
      float p1 = (v1 == -INFINITY) ? 0.f : __expf(v1 - nm);
      ldsP[(j + 8 * half) * 32 + nidx]      = (_Float16)p0;
      ldsP[(j + 8 * half) * 32 + 16 + nidx] = (_Float16)p1;

      float sum = p0 + p1;
      for (int off = 1; off < 16; off <<= 1)
        sum += __shfl_xor(sum, off, 32);

      float f = (mrow[j] == nm) ? 1.f : __expf(mrow[j] - nm);
      lrow[j] = lrow[j] * f + sum;
      mrow[j] = nm;
      for (int dt = 0; dt < 4; ++dt) o_acc[dt][j] *= f;
    }

    asm volatile("s_wait_dscnt 0" ::: "memory");
    v16h pa = load_a_tile(ldsP, 32, lane, 0);   // P as 16x32 A-tile

    for (int dt = 0; dt < 4; ++dt) {
      const _Float16* vb = vbase + (size_t)(dt * 16) * TT + t0;
      v16h bv = load_b_tile(vb, TT, lane, 0);
      o_acc[dt] = wmma_f16(pa, bv, o_acc[dt]);
    }
  }

  for (int j = 0; j < 8; ++j) {
    float inv = (lrow[j] > 0.f) ? (1.f / lrow[j]) : 0.f;   // nan_to_num
    int s = s0 + j + 8 * half;
    size_t rowbase = ((size_t)s * BB + b) * CC + (size_t)h * HC;
    for (int dt = 0; dt < 4; ++dt)
      X[rowbase + dt * 16 + nidx] = (_Float16)(o_acc[dt][j] * inv);
  }
}

// ---------------------------------------------------------------- launch
extern "C" void kernel_launch(void* const* d_in, const int* in_sizes, int n_in,
                              void* d_out, int out_size, void* d_ws, size_t ws_size,
                              hipStream_t stream) {
  const float* xq  = (const float*)d_in[0];
  const int*   pad = (const int*)d_in[1];
  const unsigned char* mask = (const unsigned char*)d_in[2];
  const int*   mlen = (const int*)d_in[3];
  const float* mkv = (const float*)d_in[4];
  const float* Wq = (const float*)d_in[5];  const float* bq = (const float*)d_in[6];
  const float* Wk = (const float*)d_in[7];  const float* bk = (const float*)d_in[8];
  const float* Wv = (const float*)d_in[9];  const float* bv = (const float*)d_in[10];
  const float* Wo = (const float*)d_in[11]; const float* bo = (const float*)d_in[12];
  float* out = (float*)d_out;

  char* ws = (char*)d_ws;
  size_t off = 0;
  auto alloc = [&](size_t bytes) {
    char* p = ws + off;
    off += (bytes + 255) & ~(size_t)255;
    return p;
  };
  _Float16* xq16 = (_Float16*)alloc((size_t)NROWS * CC * 2);
  _Float16* wq16 = (_Float16*)alloc((size_t)CC * CC * 2);
  _Float16* wk16 = (_Float16*)alloc((size_t)CC * CC * 2);
  _Float16* wv16 = (_Float16*)alloc((size_t)CC * CC * 2);
  _Float16* wo16 = (_Float16*)alloc((size_t)CC * CC * 2);
  _Float16* Q16  = (_Float16*)alloc((size_t)BB * HH * QS * HC * 2);
  _Float16* K16  = (_Float16*)alloc((size_t)BB * HH * QS * HC * 2);
  _Float16* V16  = (_Float16*)alloc((size_t)BB * HH * QS * HC * 2);
  _Float16* Kc   = (_Float16*)alloc((size_t)BB * HH * TT * HC * 2);
  _Float16* Vt   = (_Float16*)alloc((size_t)BB * HH * TT * HC * 2);
  unsigned char* cm = (unsigned char*)alloc((size_t)BB * TT * QS);
  _Float16* X16  = (_Float16*)alloc((size_t)NROWS * CC * 2);

  // f32 -> f16 conversions
  cvt_f32_f16<<<(NROWS * CC) / 256, 256, 0, stream>>>(xq, xq16, NROWS * CC);
  cvt_f32_f16<<<(CC * CC) / 256, 256, 0, stream>>>(Wq, wq16, CC * CC);
  cvt_f32_f16<<<(CC * CC) / 256, 256, 0, stream>>>(Wk, wk16, CC * CC);
  cvt_f32_f16<<<(CC * CC) / 256, 256, 0, stream>>>(Wv, wv16, CC * CC);
  cvt_f32_f16<<<(CC * CC) / 256, 256, 0, stream>>>(Wo, wo16, CC * CC);

  // Q/K/V projections: 256 blocks x (16 row-blocks, 16 col-blocks)
  gemm_nt<<<256, 256, 0, stream>>>(xq16, wq16, bq, Q16, nullptr, 0);
  gemm_nt<<<256, 256, 0, stream>>>(xq16, wk16, bk, K16, nullptr, 0);
  gemm_nt<<<256, 256, 0, stream>>>(xq16, wv16, bv, V16, nullptr, 0);

  // Combined K/V and mask
  build_kv<<<(BB * HH * TT * HC) / 256, 256, 0, stream>>>(mkv, K16, V16, mlen, Kc, Vt);
  build_mask<<<(BB * TT * QS) / 256, 256, 0, stream>>>(mask, mlen, pad, cm);

  // Flash attention: 8*16*32 = 4096 waves
  attention<<<BB * HH * (QS / 16), 32, 0, stream>>>(Q16, Kc, Vt, cm, X16);

  // Output projection -> f32 d_out
  gemm_nt<<<256, 256, 0, stream>>>(X16, wo16, bo, nullptr, out, 1);
}
